// ImproperNet_37280316130039
// MI455X (gfx1250) — compile-verified, matrix operand order checked by reading, not run on hardware
//
#include <hip/hip_runtime.h>

// ---------------------------------------------------------------------------
// ImproperNet on MI455X (gfx1250, wave32, WMMA)
//
// Per improper: E = (MLP_k(sum_j MLP_pair(r[i0]||r[ij])))^2 / 2 * phi^2
// GEMM-dominated (~15 GFLOP per call) -> v_wmma_f32_16x16x32_f16.
// One wave = 16 impropers (one WMMA M-tile). Weights live transposed in LDS
// as f16 so A and B fragments are each two ds_load_b128 per lane.
// ---------------------------------------------------------------------------

typedef __attribute__((ext_vector_type(16))) _Float16 v16h;
typedef __attribute__((ext_vector_type(8)))  float    v8f;

#define NBLOCKS          128
#define BLOCK            256
#define WAVES_PER_BLOCK  8
// LDS bytes: w1t 16384 + (w2t,w3t,k1t,k2t) 4*8192 + 388 floats (1552)
//            + 8 waves * 3072 halves (6144 B each)
#define LDS_BYTES        (16384 + 4*8192 + 1552 + WAVES_PER_BLOCK*6144)

// Wave-local LDS ordering fence: DS ops are in-order per wave; this just
// stops compiler reordering and drains DScnt before dependent reads.
__device__ __forceinline__ void lds_wave_sync() {
  asm volatile("s_wait_dscnt 0" ::: "memory");
}

// Load a 16x32 f16 fragment (A or B) from a row-major LDS buffer.
// Per ISA 16-bit A layout: element e of the v16h holds
//   K = k0 + (e&7) + 16*(e>>3) + 8*(lane>=16)
// so each lane needs two contiguous 16B chunks.
__device__ __forceinline__ v16h load_frag(const _Float16* base, int row,
                                          int stride, int k0, int hi) {
  union { v16h v; float4 q[2]; } u;
  const _Float16* p = base + row * stride + k0 + 8 * hi;
  u.q[0] = *(const float4*)(p);       // K = k0+8hi .. +7
  u.q[1] = *(const float4*)(p + 16);  // K = k0+16+8hi .. +7
  return u.v;
}

// acc[nt] covers C columns nt*16..nt*16+15. Splat bias (times scale).
__device__ __forceinline__ void init_bias(v8f acc[4], const float* b, int col,
                                          float scale) {
#pragma unroll
  for (int nt = 0; nt < 4; ++nt) {
    float bv = scale * b[nt * 16 + col];
    v8f c;
#pragma unroll
    for (int g = 0; g < 8; ++g) c[g] = bv;
    acc[nt] = c;
  }
}

// C += A(16xK from LDS, row stride astride) * W (K x 64, stored transposed
// in LDS as WT[64][K], row-major in K).
__device__ __forceinline__ void gemm16(const _Float16* A, int astride,
                                       const _Float16* WT, int K,
                                       v8f acc[4], int col, int hi) {
  const int nk = K >> 5;
#pragma unroll
  for (int kc = 0; kc < 4; ++kc) {
    if (kc >= nk) break;
    v16h a = load_frag(A, col, astride, kc * 32, hi);
#pragma unroll
    for (int nt = 0; nt < 4; ++nt) {
      v16h b = load_frag(WT, nt * 16 + col, K, kc * 32, hi);
      acc[nt] = __builtin_amdgcn_wmma_f32_16x16x32_f16(
          false, a, false, b, (short)0, acc[nt], false, false);
    }
  }
}

// C layout: VGPR g at lane l -> M = g + 8*(l>=16), N = (l&15) within tile nt.
__device__ __forceinline__ void store_acc(_Float16* H, const v8f acc[4],
                                          int col, int hi, bool doTanh) {
#pragma unroll
  for (int nt = 0; nt < 4; ++nt) {
#pragma unroll
    for (int g = 0; g < 8; ++g) {
      float v = acc[nt][g];
      if (doTanh) v = tanhf(v);
      H[(g + 8 * hi) * 64 + nt * 16 + col] = (_Float16)v;
    }
  }
}

__global__ void __launch_bounds__(BLOCK) improper_zero(float* o, int n) {
  int i = blockIdx.x * blockDim.x + threadIdx.x;
  if (i < n) o[i] = 0.0f;
}

__global__ void __launch_bounds__(BLOCK) improper_main(
    const float* __restrict__ r,   const float* __restrict__ xyz,
    const int* __restrict__ impropers, const int* __restrict__ segIds,
    const float* __restrict__ nw1, const float* __restrict__ nb1,
    const float* __restrict__ nw2, const float* __restrict__ nb2,
    const float* __restrict__ nw3, const float* __restrict__ nb3,
    const float* __restrict__ kw1, const float* __restrict__ kb1,
    const float* __restrict__ kw2, const float* __restrict__ kb2,
    const float* __restrict__ kw3, const float* __restrict__ kb3,
    float* __restrict__ out, int numTiles) {
  extern __shared__ char smem[];
  _Float16* w1t = (_Float16*)smem;        // [64][128]  Wᵀ for layer1
  _Float16* w2t = w1t + 64 * 128;         // [64][64]
  _Float16* w3t = w2t + 64 * 64;
  _Float16* k1t = w3t + 64 * 64;
  _Float16* k2t = k1t + 64 * 64;
  float* fl   = (float*)(k2t + 64 * 64);
  float* kw3s = fl;                        // [64]
  float* nb1f = fl + 64;
  float* nb2f = fl + 128;
  float* nb3f = fl + 192;
  float* kb1f = fl + 256;
  float* kb2f = fl + 320;                  // fl[384] = kb3
  _Float16* waveMem = (_Float16*)(fl + 388);

  const int tid = threadIdx.x;

  // ---- Stage weights transposed to f16 (once per block) ----
  for (int e = tid; e < 128 * 64; e += BLOCK) {
    int k = e >> 6, n = e & 63;
    w1t[n * 128 + k] = (_Float16)nw1[e];
  }
  for (int e = tid; e < 64 * 64; e += BLOCK) {
    int k = e >> 6, n = e & 63;
    w2t[n * 64 + k] = (_Float16)nw2[e];
    w3t[n * 64 + k] = (_Float16)nw3[e];
    k1t[n * 64 + k] = (_Float16)kw1[e];
    k2t[n * 64 + k] = (_Float16)kw2[e];
  }
  if (tid < 64) {
    kw3s[tid] = kw3[tid];
    nb1f[tid] = nb1[tid];  nb2f[tid] = nb2[tid];  nb3f[tid] = nb3[tid];
    kb1f[tid] = kb1[tid];  kb2f[tid] = kb2[tid];
    if (tid == 0) fl[384] = kb3[0];
  }
  __syncthreads();

  const int lane   = tid & 31;
  const int waveId = tid >> 5;
  const int col    = lane & 15;   // = improper row within tile for A, column N for B and C
  const int hi     = lane >> 4;
  _Float16* X = waveMem + waveId * 3072;  // 16x128 f16 staging
  _Float16* H = X + 2048;                 // 16x64  f16 staging

  const int slot = blockIdx.x * WAVES_PER_BLOCK + waveId;
  const int step = gridDim.x * WAVES_PER_BLOCK;

  for (int t = slot; t < numTiles; t += step) {
    const int m = t * 16 + col;
    const int4 idx4 = ((const int4*)impropers)[m];

    // ---- geometry: phi (lanes 0..15, one improper each) ----
    float phi = 0.0f;
    if (hi == 0) {
      const float* p0 = xyz + 3 * (size_t)idx4.x;
      const float* p1 = xyz + 3 * (size_t)idx4.y;
      const float* p2 = xyz + 3 * (size_t)idx4.z;
      const float* p3 = xyz + 3 * (size_t)idx4.w;
      float v1x = p1[0]-p0[0], v1y = p1[1]-p0[1], v1z = p1[2]-p0[2];
      float v2x = p1[0]-p2[0], v2y = p1[1]-p2[1], v2z = p1[2]-p2[2];
      float v3x = -v2x,        v3y = -v2y,        v3z = -v2z;
      float v4x = p2[0]-p3[0], v4y = p2[1]-p3[1], v4z = p2[2]-p3[2];
      float c1x = v1y*v2z - v1z*v2y, c1y = v1z*v2x - v1x*v2z, c1z = v1x*v2y - v1y*v2x;
      float c2x = v3y*v4z - v3z*v4y, c2y = v3z*v4x - v3x*v4z, c2z = v3x*v4y - v3y*v4x;
      float n1 = c1x*c1x + c1y*c1y + c1z*c1z;
      float n2 = c2x*c2x + c2y*c2y + c2z*c2z;
      float cosphi = (c1x*c2x + c1y*c2y + c1z*c2z) * rsqrtf(n1 * n2);
      phi = acosf(cosphi / 1.000001f);
    }

    // improper_input accumulator; each pair MLP adds b3 once -> init 3*b3
    v8f accSum[4];
    init_bias(accSum, nb3f, col, 3.0f);

    // ---- 3 pair MLPs ----
#pragma unroll
    for (int j = 0; j < 3; ++j) {
      const int ij   = (j == 0) ? idx4.y : (j == 1) ? idx4.z : idx4.w;
      const int gidx = hi ? ij : idx4.x;
      // build X[16][128] = concat(r[i0] f16, r[ij] f16); lane -> (row=col, half=hi)
      {
        const float4* src = (const float4*)(r + 64 * (size_t)gidx);
        uint2* dst = (uint2*)(X + col * 128 + hi * 64);
#pragma unroll
        for (int c = 0; c < 16; ++c) {
          float4 f = src[c];
          union { _Float16 h[4]; uint2 u; } pk;
          pk.h[0] = (_Float16)f.x; pk.h[1] = (_Float16)f.y;
          pk.h[2] = (_Float16)f.z; pk.h[3] = (_Float16)f.w;
          dst[c] = pk.u;
        }
      }
      lds_wave_sync();

      v8f acc[4];
      init_bias(acc, nb1f, col, 1.0f);
      gemm16(X, 128, w1t, 128, acc, col, hi);       // layer1: 16x128 @ 128x64
      store_acc(H, acc, col, hi, true);             // tanh -> H
      lds_wave_sync();

      init_bias(acc, nb2f, col, 1.0f);
      gemm16(H, 64, w2t, 64, acc, col, hi);         // layer2
      store_acc(X, acc, col, hi, true);             // tanh -> reuse X as H2
      lds_wave_sync();

      gemm16(X, 64, w3t, 64, accSum, col, hi);      // layer3, C chained
    }

    // ---- k MLP ----
    store_acc(H, accSum, col, hi, false);           // improper_input -> f16
    lds_wave_sync();

    v8f acc[4];
    init_bias(acc, kb1f, col, 1.0f);
    gemm16(H, 64, k1t, 64, acc, col, hi);
    store_acc(X, acc, col, hi, true);
    lds_wave_sync();

    init_bias(acc, kb2f, col, 1.0f);
    gemm16(X, 64, k2t, 64, acc, col, hi);
    store_acc(H, acc, col, hi, true);
    lds_wave_sync();

    // ---- final 64->1 dot, E, segment reduce (lanes 0..15) ----
    if (hi == 0) {
      float kv = fl[384];  // kb3
      const _Float16* hrow = H + col * 64;
      for (int c = 0; c < 64; ++c) kv += (float)hrow[c] * kw3s[c];
      float E = 0.5f * kv * kv * phi * phi;
#pragma unroll
      for (int off = 8; off > 0; off >>= 1) E += __shfl_down(E, off, 16);
      if (lane == 0) atomicAdd(out + segIds[t * 16], E);
    }
  }
}

extern "C" void kernel_launch(void* const* d_in, const int* in_sizes, int n_in,
                              void* d_out, int out_size, void* d_ws, size_t ws_size,
                              hipStream_t stream) {
  const float* r   = (const float*)d_in[0];
  const float* xyz = (const float*)d_in[1];
  const int* impropers = (const int*)d_in[2];
  const int* segIds    = (const int*)d_in[3];
  const float* nw1 = (const float*)d_in[4];
  const float* nb1 = (const float*)d_in[5];
  const float* nw2 = (const float*)d_in[6];
  const float* nb2 = (const float*)d_in[7];
  const float* nw3 = (const float*)d_in[8];
  const float* nb3 = (const float*)d_in[9];
  const float* kw1 = (const float*)d_in[10];
  const float* kb1 = (const float*)d_in[11];
  const float* kw2 = (const float*)d_in[12];
  const float* kb2 = (const float*)d_in[13];
  const float* kw3 = (const float*)d_in[14];
  const float* kb3 = (const float*)d_in[15];
  float* out = (float*)d_out;

  const int M = in_sizes[3];       // segment_ids count == #impropers
  const int numTiles = M / 16;

  static bool attrDone = false;
  (void)attrDone;
  hipFuncSetAttribute((const void*)improper_main,
                      hipFuncAttributeMaxDynamicSharedMemorySize, LDS_BYTES);

  improper_zero<<<1, BLOCK, 0, stream>>>(out, out_size);
  improper_main<<<NBLOCKS, BLOCK, LDS_BYTES, stream>>>(
      r, xyz, impropers, segIds, nw1, nb1, nw2, nb2, nw3, nb3,
      kw1, kb1, kw2, kb2, kw3, kb3, out, numTiles);
}
